// Attention_61340722921528
// MI455X (gfx1250) — matrix operand order repaired and checked by reading, not compiled, not run on hardware
//
#include <hip/hip_runtime.h>
#include <hip/hip_bf16.h>
#include <math.h>

// ---------------------------------------------------------------------------
// Problem constants (from reference)
// ---------------------------------------------------------------------------
#define BSZ      4
#define SEQ      1024
#define HIDDEN   1024
#define NHEADS   16
#define HEADDIM  64
#define NTOK     (BSZ * SEQ)            // 4096 rows for the projections
#define NCHUNK   (BSZ * NHEADS)         // 64 independent (1024 x 64) attentions
#define SCALE_F  0.125f                 // 64^-0.5, folded into Q projection

typedef __attribute__((ext_vector_type(16))) _Float16 v16h;
typedef __attribute__((ext_vector_type(8)))  _Float16 v8h;
typedef __attribute__((ext_vector_type(8)))  float    v8f;
typedef __attribute__((ext_vector_type(4)))  unsigned int v4u;
typedef __attribute__((ext_vector_type(8)))  int      v8i;
typedef __attribute__((ext_vector_type(4)))  int      v4i;

#if defined(__has_builtin)
#if __has_builtin(__builtin_amdgcn_tensor_load_to_lds) && \
    __has_builtin(__builtin_amdgcn_s_wait_tensorcnt)
#define USE_TDM 1
#endif
#if __has_builtin(__builtin_amdgcn_permlane16)
#define USE_PERMLANE 1
#endif
#endif
#ifndef USE_TDM
#define USE_TDM 0
#endif
#ifndef USE_PERMLANE
#define USE_PERMLANE 0
#endif

// Build a 16-element f16 fragment from two 16-byte chunks (b128 loads).
__device__ __forceinline__ v16h make_frag(const _Float16* lo, const _Float16* hi) {
    v8h a = *(const v8h*)lo;
    v8h b = *(const v8h*)hi;
    return __builtin_shufflevector(a, b, 0,1,2,3,4,5,6,7,8,9,10,11,12,13,14,15);
}

__device__ __forceinline__ v8f wmma16(v16h a, v16h b, v8f c) {
    // D = A(16x32) * B(32x16) + C, fp32 accumulate
    return __builtin_amdgcn_wmma_f32_16x16x32_f16(false, a, false, b, (short)0, c,
                                                  false, false);
}

// max over the 16 lanes of each wave32 half, via VALU permlane (no LDS pipe).
__device__ __forceinline__ float rowmax16(float v) {
#if USE_PERMLANE
    // nibble i of {hi,lo} selects the source row-lane for row position i
#define XORMAX16(LO, HI)                                                      \
    {                                                                         \
        unsigned _x = __float_as_uint(v);                                     \
        unsigned _y = __builtin_amdgcn_permlane16(_x, _x, (LO), (HI),         \
                                                  false, false);              \
        v = fmaxf(v, __uint_as_float(_y));                                    \
    }
    XORMAX16(0x67452301u, 0xefcdab89u)   // xor 1
    XORMAX16(0x54761032u, 0xdcfe98bau)   // xor 2
    XORMAX16(0x32107654u, 0xba98fedcu)   // xor 4
    XORMAX16(0xfedcba98u, 0x76543210u)   // xor 8
#undef XORMAX16
#else
#pragma unroll
    for (int off = 1; off < 16; off <<= 1)
        v = fmaxf(v, __shfl_xor(v, off, 32));
#endif
    return v;
}

#if USE_TDM
// Truncate a generic pointer to a shared-memory object down to its LDS byte
// offset (flat LDS aperture keeps the offset in addr[31:0]).
__device__ __forceinline__ unsigned lds_byte_off(const void* p) {
    return (unsigned)(unsigned long long)(uintptr_t)p;
}

// Issue a 2D TDM load: tile_d1 rows x tile_d0 f16 elements, row stride
// `stride0` elements in global memory, destination LDS rows padded by
// 16 bytes after every 64 bytes (i.e. 32 f16 data + 8 f16 pad = 40-half rows).
// D# field packing per CDNA5 ISA 8.3/8.4 (group0 128b, group1 256b).
__device__ __forceinline__ void tdm_load_2d(unsigned lds_off, const void* gsrc,
                                            unsigned tile_d0, unsigned tile_d1,
                                            unsigned long long stride0) {
    unsigned long long ga = (unsigned long long)(uintptr_t)gsrc;  // 57-bit VA
    // ---- group 0 ----
    unsigned long long g0lo = 1ull                       // count = 1 valid D#
                            | ((unsigned long long)lds_off << 32);
    unsigned long long g0hi = ga | (2ull << 62);         // type = 2 ("image")
    v4u g0 = { (unsigned)g0lo, (unsigned)(g0lo >> 32),
               (unsigned)g0hi, (unsigned)(g0hi >> 32) };
    // ---- group 1 ----
    const unsigned long long data_size = 1;   // 2-byte elements
    const unsigned long long pad_en    = 1;   // pad LDS rows
    const unsigned long long pad_int   = 3;   // after 16 DWORDs (64B)
    const unsigned long long pad_amt   = 3;   // add 4 DWORDs (16B)
    unsigned long long q0 = (data_size << 16) | (pad_en << 20)
                          | (pad_int << 22)   | (pad_amt << 25)
                          | ((unsigned long long)(tile_d0 & 0xFFFFu) << 48); // tensor_dim0
    unsigned long long q1 = 0ull                                   // tensor_dim0 hi bits
                          | ((unsigned long long)tile_d1 << 16)    // tensor_dim1
                          | ((unsigned long long)tile_d0 << 48);   // tile_dim0
    unsigned long long q2 = (unsigned long long)tile_d1            // tile_dim1
                          | (0ull << 16)                           // tile_dim2 unused
                          | ((stride0 & 0xFFFFFFFFull) << 32);     // dim0 stride lo
    unsigned long long q3 = ((stride0 >> 32) & 0xFFFFull);         // dim0 stride hi
    v8i g1 = { (int)q0, (int)(q0 >> 32), (int)q1, (int)(q1 >> 32),
               (int)q2, (int)(q2 >> 32), (int)q3, (int)(q3 >> 32) };
    v4i gz4 = { 0, 0, 0, 0 };          // groups 2/3: zero (2D tile)
    v8i gz8 = { 0, 0, 0, 0, 0, 0, 0, 0 };
    // 6-arg toolchain form: (g0, g1, g2, g3, extra, cpol)
    __builtin_amdgcn_tensor_load_to_lds(g0, g1, gz4, gz4, gz8, 0);
}
#endif

// ---------------------------------------------------------------------------
// fp32 -> f16 conversion (vectorized by 4)
// ---------------------------------------------------------------------------
__global__ void cvt_f32_to_f16(const float* __restrict__ src,
                               _Float16* __restrict__ dst, int n4) {
    int i = blockIdx.x * blockDim.x + threadIdx.x;
    if (i < n4) {
        float4 v = ((const float4*)src)[i];
        struct __align__(8) H4 { _Float16 a, b, c, d; };
        ((H4*)dst)[i] = H4{ (_Float16)v.x, (_Float16)v.y,
                            (_Float16)v.z, (_Float16)v.w };
    }
}

// ---------------------------------------------------------------------------
// GEMM: C[M x N] = cscale * A[M x K] * B[N x K]^T  (A, B f16 row-major)
// Block tile 128x128, 8 waves (4 row-groups x 2 col-groups), wave tile 32x64.
// TDM path: per K-step the 128x32 A and B slabs are DMA'd into padded LDS
// (double-buffered, DMA of step k+32 overlaps WMMAs of step k); fragments come
// from ds_read_b128. Fallback path loads fragments straight from global.
// ---------------------------------------------------------------------------
template <typename OutT>
__global__ __launch_bounds__(256) void gemm_f16_nt(const _Float16* __restrict__ A,
                                                   const _Float16* __restrict__ B,
                                                   OutT* __restrict__ C,
                                                   int M, int N, int K,
                                                   float cscale) {
    const int lane = threadIdx.x & 31;
    const int w    = threadIdx.x >> 5;
    const int wr   = w & 3;           // wave row group (0..3)
    const int wc   = w >> 2;          // wave col group (0..1)
    const int lm   = lane & 15;
    const int half = lane >> 4;
    const int aoff = half ? 8 : 0;    // A-operand per-lane K chunk start
    const int boff = half ? 8 : 0;    // B-operand per-lane K chunk start

    const int r0 = blockIdx.y * 128 + wr * 32;
    const int c0 = blockIdx.x * 128 + wc * 64;

    v8f acc[2][4];
#pragma unroll
    for (int i = 0; i < 2; ++i)
#pragma unroll
        for (int j = 0; j < 4; ++j) acc[i][j] = (v8f)0.0f;

#if USE_TDM
    __shared__ _Float16 As[2][128][40];   // 32 data + 8 pad halves per row
    __shared__ _Float16 Bs[2][128][40];

    const _Float16* Ablk = A + (size_t)(blockIdx.y * 128) * K;
    const _Float16* Bblk = B + (size_t)(blockIdx.x * 128) * K;

    if (w == 0) {   // one wave drives the tensor DMA (EXEC-independent)
        tdm_load_2d(lds_byte_off(&As[0][0][0]), Ablk, 32, 128, (unsigned long long)K);
        tdm_load_2d(lds_byte_off(&Bs[0][0][0]), Bblk, 32, 128, (unsigned long long)K);
    }

    for (int k0 = 0; k0 < K; k0 += 32) {
        const int buf = (k0 >> 5) & 1;
        if (w == 0) __builtin_amdgcn_s_wait_tensorcnt(0);
        __syncthreads();              // publish DMA'd slabs to all waves
        if (w == 0 && k0 + 32 < K) {  // prefetch next K-step while we compute
            tdm_load_2d(lds_byte_off(&As[buf ^ 1][0][0]), Ablk + (k0 + 32),
                        32, 128, (unsigned long long)K);
            tdm_load_2d(lds_byte_off(&Bs[buf ^ 1][0][0]), Bblk + (k0 + 32),
                        32, 128, (unsigned long long)K);
        }

        v16h af[2], bf[4];
#pragma unroll
        for (int i = 0; i < 2; ++i) {
            const _Float16* ap = &As[buf][wr * 32 + i * 16 + lm][aoff];
            af[i] = make_frag(ap, ap + 16);
        }
#pragma unroll
        for (int j = 0; j < 4; ++j) {
            const _Float16* bp = &Bs[buf][wc * 64 + j * 16 + lm][boff];
            bf[j] = make_frag(bp, bp + 16);
        }
#pragma unroll
        for (int i = 0; i < 2; ++i)
#pragma unroll
            for (int j = 0; j < 4; ++j)
                acc[i][j] = wmma16(af[i], bf[j], acc[i][j]);
    }
#else
    for (int k0 = 0; k0 < K; k0 += 32) {
        v16h af[2], bf[4];
#pragma unroll
        for (int i = 0; i < 2; ++i) {
            const _Float16* ap = A + (size_t)(r0 + i * 16 + lm) * K + k0 + aoff;
            af[i] = make_frag(ap, ap + 16);
        }
#pragma unroll
        for (int j = 0; j < 4; ++j) {
            const _Float16* bp = B + (size_t)(c0 + j * 16 + lm) * K + k0 + boff;
            bf[j] = make_frag(bp, bp + 16);
        }
#pragma unroll
        for (int i = 0; i < 2; ++i)
#pragma unroll
            for (int j = 0; j < 4; ++j)
                acc[i][j] = wmma16(af[i], bf[j], acc[i][j]);
    }
#endif

    // C layout: lane holds column lm; VGPR e -> row e (lanes 0-15) / e+8 (16-31)
#pragma unroll
    for (int i = 0; i < 2; ++i)
#pragma unroll
        for (int j = 0; j < 4; ++j)
#pragma unroll
            for (int e = 0; e < 8; ++e) {
                int row = r0 + i * 16 + e + 8 * half;
                int col = c0 + j * 16 + lm;
                C[(size_t)row * N + col] = (OutT)(acc[i][j][e] * cscale);
            }
}

// ---------------------------------------------------------------------------
// Flash attention over 64 independent (SEQ x HEADDIM) slabs.
// grid = (SEQ/128, 64). Block = 256 threads = 8 waves; wave owns 16 full rows.
// Q is pre-scaled by 1/sqrt(d) (exact in f16), so the WMMA outputs are the
// final logits. Row-max via permlane16 butterfly (VALU); row-sum via WMMA
// against a ones fragment (result lands in C layout: each lane already holds
// its 8 rows' sums -> zero cross-lane traffic).
// ---------------------------------------------------------------------------
__global__ __launch_bounds__(256) void attn_flash(const _Float16* __restrict__ Q,
                                                  const _Float16* __restrict__ K,
                                                  const _Float16* __restrict__ V,
                                                  _Float16* __restrict__ O) {
    const size_t chunk = (size_t)blockIdx.y * SEQ * HEADDIM;
    const _Float16* Qc = Q + chunk;
    const _Float16* Kc = K + chunk;
    const _Float16* Vc = V + chunk;
    _Float16*       Oc = O + chunk;

    const int lane = threadIdx.x & 31;
    const int w    = threadIdx.x >> 5;
    const int lm   = lane & 15;
    const int half = lane >> 4;
    const int aoff = half ? 8 : 0;
    const int boff = half ? 8 : 0;
    const int r0   = blockIdx.x * 128 + w * 16;   // this wave's 16 rows

    __shared__ _Float16 Vt[HEADDIM][40];          // V^T staging (padded rows)
    __shared__ _Float16 Pst[8][16][40];           // per-wave P staging

    // Q fragments for the two d-steps (d = 0..31, 32..63), kept in registers
    v16h qa[2];
#pragma unroll
    for (int ds = 0; ds < 2; ++ds) {
        const _Float16* qp = Qc + (size_t)(r0 + lm) * HEADDIM + ds * 32 + aoff;
        qa[ds] = make_frag(qp, qp + 16);
    }

    v16h vones;
#pragma unroll
    for (int e = 0; e < 16; ++e) vones[e] = (_Float16)1.0f;

    v8f oacc[4];
#pragma unroll
    for (int ct = 0; ct < 4; ++ct) oacc[ct] = (v8f)0.0f;
    float mrun[8], lrun[8];
#pragma unroll
    for (int e = 0; e < 8; ++e) { mrun[e] = -INFINITY; lrun[e] = 0.0f; }

    // this thread's V-staging assignment: one v8h global load per iteration
    const int vj = threadIdx.x >> 3;         // 0..31 (j row)
    const int vd = (threadIdx.x & 7) * 8;    // d chunk of 8

    for (int jb = 0; jb < SEQ; jb += 32) {
        __syncthreads();   // all waves done reading Vt from previous iteration
        {   // cooperative transposed staging of the 32x64 V block
            v8h vv = *(const v8h*)(Vc + (size_t)(jb + vj) * HEADDIM + vd);
#pragma unroll
            for (int q = 0; q < 8; ++q) Vt[vd + q][vj] = vv[q];
        }
        __syncthreads();

        // S = Q K^T for 2 column tiles of 16 (K rows are row-major = B layout)
        v8f sc[2];
        sc[0] = (v8f)0.0f; sc[1] = (v8f)0.0f;
#pragma unroll
        for (int t = 0; t < 2; ++t)
#pragma unroll
            for (int ds = 0; ds < 2; ++ds) {
                const _Float16* kp =
                    Kc + (size_t)(jb + t * 16 + lm) * HEADDIM + ds * 32 + boff;
                v16h kb = make_frag(kp, kp + 16);
                sc[t] = wmma16(qa[ds], kb, sc[t]);
            }

        // online softmax max-track; each lane carries 8 rows (e + 8*half)
        float alpha[8];
#pragma unroll
        for (int e = 0; e < 8; ++e) {
            float s0 = sc[0][e];
            float s1 = sc[1][e];
            float v = rowmax16(fmaxf(s0, s1));
            float mnew = fmaxf(mrun[e], v);
            alpha[e] = __expf(mrun[e] - mnew);
            mrun[e] = mnew;
            sc[0][e] = __expf(s0 - mnew);
            sc[1][e] = __expf(s1 - mnew);
            lrun[e] *= alpha[e];          // row-sum added below via WMMA
        }
#pragma unroll
        for (int ct = 0; ct < 4; ++ct)
#pragma unroll
            for (int e = 0; e < 8; ++e) oacc[ct][e] *= alpha[e];

        // P: C-layout -> LDS -> A-layout fragment
#pragma unroll
        for (int t = 0; t < 2; ++t)
#pragma unroll
            for (int e = 0; e < 8; ++e)
                Pst[w][e + 8 * half][t * 16 + lm] = (_Float16)sc[t][e];
        asm volatile("s_wait_dscnt 0" ::: "memory");   // wave-local LDS RAW fence

        const _Float16* pp = &Pst[w][lm][aoff];
        v16h pa = make_frag(pp, pp + 16);

        // row-sum of P via WMMA against ones: rs lands in C layout
        v8f rs = (v8f)0.0f;
        rs = wmma16(pa, vones, rs);
#pragma unroll
        for (int e = 0; e < 8; ++e) lrun[e] += rs[e];

#pragma unroll
        for (int ct = 0; ct < 4; ++ct) {
            const _Float16* vp = &Vt[ct * 16 + lm][boff];
            v16h vb = make_frag(vp, vp + 16);
            oacc[ct] = wmma16(pa, vb, oacc[ct]);
        }
    }

    float invl[8];
#pragma unroll
    for (int e = 0; e < 8; ++e) invl[e] = 1.0f / lrun[e];
#pragma unroll
    for (int ct = 0; ct < 4; ++ct)
#pragma unroll
        for (int e = 0; e < 8; ++e) {
            int row = r0 + e + 8 * half;
            int d   = ct * 16 + lm;
            Oc[(size_t)row * HEADDIM + d] = (_Float16)(oacc[ct][e] * invl[e]);
        }
}

// ---------------------------------------------------------------------------
// Launcher
// ---------------------------------------------------------------------------
extern "C" void kernel_launch(void* const* d_in, const int* in_sizes, int n_in,
                              void* d_out, int out_size, void* d_ws, size_t ws_size,
                              hipStream_t stream) {
    (void)in_sizes; (void)n_in; (void)out_size; (void)ws_size;
    const float* x  = (const float*)d_in[0];
    const float* wq = (const float*)d_in[1];
    const float* wk = (const float*)d_in[2];
    const float* wv = (const float*)d_in[3];
    const float* wo = (const float*)d_in[4];
    float* out = (float*)d_out;

    const size_t NX = (size_t)NTOK * HIDDEN;      // 4M
    const size_t NW = (size_t)HIDDEN * HIDDEN;    // 1M

    _Float16* ws = (_Float16*)d_ws;
    _Float16* xh  = ws;                 // 4M halves
    _Float16* wqh = xh  + NX;
    _Float16* wkh = wqh + NW;
    _Float16* wvh = wkh + NW;
    _Float16* woh = wvh + NW;
    _Float16* qh  = woh + NW;           // 4M
    _Float16* kh  = qh  + NX;
    _Float16* vh  = kh  + NX;
    _Float16* ah  = vh  + NX;           // attention output, 4M (total 48 MB)

    // 1) convert inputs to f16
    {
        int n4 = (int)(NX / 4);
        cvt_f32_to_f16<<<(n4 + 255) / 256, 256, 0, stream>>>(x, xh, n4);
        n4 = (int)(NW / 4);
        cvt_f32_to_f16<<<(n4 + 255) / 256, 256, 0, stream>>>(wq, wqh, n4);
        cvt_f32_to_f16<<<(n4 + 255) / 256, 256, 0, stream>>>(wk, wkh, n4);
        cvt_f32_to_f16<<<(n4 + 255) / 256, 256, 0, stream>>>(wv, wvh, n4);
        cvt_f32_to_f16<<<(n4 + 255) / 256, 256, 0, stream>>>(wo, woh, n4);
    }

    // 2) QKV projections: C[4096 x 1024] = X * W^T, f16 output.
    //    Q is pre-scaled by 1/sqrt(64) = 0.125 (exact exponent shift in f16).
    dim3 gproj(HIDDEN / 128, NTOK / 128);  // (8, 32)
    gemm_f16_nt<_Float16><<<gproj, 256, 0, stream>>>(xh, wqh, qh, NTOK, HIDDEN,
                                                     HIDDEN, SCALE_F);
    gemm_f16_nt<_Float16><<<gproj, 256, 0, stream>>>(xh, wkh, kh, NTOK, HIDDEN,
                                                     HIDDEN, 1.0f);
    gemm_f16_nt<_Float16><<<gproj, 256, 0, stream>>>(xh, wvh, vh, NTOK, HIDDEN,
                                                     HIDDEN, 1.0f);

    // 3) flash attention over the 64 raw-view slabs
    attn_flash<<<dim3(SEQ / 128, NCHUNK), 256, 0, stream>>>(qh, kh, vh, ah);

    // 4) output projection: C[4096 x 1024] = attn * Wo^T, f32 output to d_out
    gemm_f16_nt<float><<<gproj, 256, 0, stream>>>(ah, woh, out, NTOK, HIDDEN,
                                                  HIDDEN, 1.0f);
}
